// AnticipatoryTransformer_23545010716769
// MI455X (gfx1250) — matrix operand order repaired and checked by reading, not compiled, not run on hardware
//
#include <hip/hip_runtime.h>
#include <hip/hip_bf16.h>
#include <math.h>

// ---------------------------------------------------------------------------
// AnticipatoryTransformer forward for MI455X (gfx1250, wave32, WMMA)
// ---------------------------------------------------------------------------

typedef _Float16 half_t;
typedef __attribute__((ext_vector_type(16))) _Float16 v16h;
typedef __attribute__((ext_vector_type(8)))  _Float16 v8h;
typedef __attribute__((ext_vector_type(4)))  _Float16 v4h;
typedef __attribute__((ext_vector_type(8)))  float    v8f;

#define B_  2
#define S_  1024
#define D_  1024
#define H_  16
#define DH_ 64
#define L_  4
#define FF_ 4096
#define NEG_ (-1e9f)

// ---------------------------------------------------------------------------
// WMMA fragment helpers (wave32 layouts per CDNA5 ISA 7.12.2)
// ---------------------------------------------------------------------------

__device__ __forceinline__ v16h cat8(v8h a, v8h b) {
  return __builtin_shufflevector(a, b, 0, 1, 2, 3, 4, 5, 6, 7,
                                 8, 9, 10, 11, 12, 13, 14, 15);
}

// A matrix 16x32 f16: lane l holds row m=l&15; halves j: k = (j/8)*16 + kb + j%8,
// kb = (l<16)?0:8  -> two contiguous 8-half groups.
__device__ __forceinline__ v16h load_a_frag(const half_t* __restrict__ p, int ld) {
  int l  = threadIdx.x & 31;
  int m  = l & 15;
  int kb = (l < 16) ? 0 : 8;
  const half_t* row = p + (long)m * ld;
  return cat8(*(const v8h*)(row + kb), *(const v8h*)(row + 16 + kb));
}

// B matrix 32x16 f16 where the "k" dimension is CONTIGUOUS per row of p:
// lane l holds column n=l&15 (a row of p), halves j: k = (j/8)*16 + kb + j%8, kb=(l<16)?0:8.
// Used for K^T, V^T, and the transposed weight tiles in LDS.
__device__ __forceinline__ v16h load_b_frag_rows(const half_t* __restrict__ p, int ld) {
  int l = threadIdx.x & 31;
  const half_t* q = p + (long)(l & 15) * ld + ((l < 16) ? 0 : 8);
  return cat8(*(const v8h*)(q), *(const v8h*)(q + 16));
}

__device__ __forceinline__ float redmax16(float v) {
#pragma unroll
  for (int m = 1; m < 16; m <<= 1) v = fmaxf(v, __shfl_xor(v, m, 32));
  return v;
}
__device__ __forceinline__ float redsum16(float v) {
#pragma unroll
  for (int m = 1; m < 16; m <<= 1) v += __shfl_xor(v, m, 32);
  return v;
}
__device__ __forceinline__ float sigm(float x) { return 1.0f / (1.0f + __expf(-x)); }
__device__ __forceinline__ float gelu_f(float x) {
  return 0.5f * x * (1.0f + erff(x * 0.70710678118654752f));
}

// ---------------------------------------------------------------------------
// Fused f32 -> f16 convert + transpose:  in[K,N] f32  ->  out[N,K] f16
// 32x32 tiles bounced through LDS; coalesced on both sides.
// ---------------------------------------------------------------------------
#define TPITCH 36  // halves; 72B rows, 8B aligned for v4h

__global__ __launch_bounds__(256) void cvtT_kernel(const float* __restrict__ in,
                                                   half_t* __restrict__ out,
                                                   int K, int N) {
  __shared__ half_t tile[32 * TPITCH];
  int k0 = blockIdx.y * 32, n0 = blockIdx.x * 32;
  int row = threadIdx.x >> 3;          // 0..31
  int c4  = (threadIdx.x & 7) * 4;     // 0..28
  {
    float4 f = *(const float4*)&in[(long)(k0 + row) * N + n0 + c4];
    v4h h;
    h[0] = (half_t)f.x; h[1] = (half_t)f.y; h[2] = (half_t)f.z; h[3] = (half_t)f.w;
    *(v4h*)&tile[row * TPITCH + c4] = h;
  }
  __syncthreads();
  v4h o;
#pragma unroll
  for (int i = 0; i < 4; ++i) o[i] = tile[(c4 + i) * TPITCH + row];
  *(v4h*)&out[(long)(n0 + row) * K + k0 + c4] = o;
}

// ---------------------------------------------------------------------------
// LayerNorm over D=1024; one row per block; f16 output
// ---------------------------------------------------------------------------
__global__ __launch_bounds__(256) void ln_kernel(const float* __restrict__ in,
                                                 const float* __restrict__ g,
                                                 const float* __restrict__ bta,
                                                 half_t* __restrict__ out) {
  __shared__ float bufs[8], bufss[8];
  int row = blockIdx.x;
  const float* x = in + (long)row * D_;
  float s = 0.f, ss = 0.f;
  for (int i = threadIdx.x; i < D_; i += 256) { float v = x[i]; s += v; ss += v * v; }
#pragma unroll
  for (int m = 1; m < 32; m <<= 1) { s += __shfl_xor(s, m, 32); ss += __shfl_xor(ss, m, 32); }
  if ((threadIdx.x & 31) == 0) { bufs[threadIdx.x >> 5] = s; bufss[threadIdx.x >> 5] = ss; }
  __syncthreads();
  if (threadIdx.x == 0) {
    float t = 0.f, tt = 0.f;
    for (int i = 0; i < 8; ++i) { t += bufs[i]; tt += bufss[i]; }
    float mean = t * (1.0f / D_);
    float var  = tt * (1.0f / D_) - mean * mean;
    bufs[0]  = mean;
    bufss[0] = rsqrtf(var + 1e-5f);
  }
  __syncthreads();
  float mean = bufs[0], rstd = bufss[0];
  for (int i = threadIdx.x; i < D_; i += 256)
    out[(long)row * D_ + i] = (half_t)((x[i] - mean) * rstd * g[i] + bta[i]);
}

// ---------------------------------------------------------------------------
// Tiled WMMA GEMM: C[M,N] = act(A[M,K] * Bt[N,K]^T + bias + residual)
// Block = 256 threads (8 waves). Tile 64x64, K step 32.
// Both A and Bt are staged as 64x32 f16 LDS tiles; all fragment reads are
// two aligned ds_load_b128. Epilogue fully specialized at compile time.
// ---------------------------------------------------------------------------
#define BM 64
#define BN 64
#define BK 32
#define APITCH 40  // halves; 80B rows -> 16B aligned

template <bool HAS_BIAS, bool HAS_RES, bool GELU, bool OUTF, bool OUTH>
__global__ __launch_bounds__(256) void gemm_kernel(
    const half_t* __restrict__ A, int lda,
    const half_t* __restrict__ Bt, int ldbt,  // [N,K] row-major (pre-transposed)
    const float*  __restrict__ bias,
    const float*  __restrict__ resid, int ldr,
    float*        __restrict__ outF, int ldo,
    half_t*       __restrict__ outH, int ldoh,
    int M, int N, int K) {
  __shared__ half_t As[BM * APITCH];
  __shared__ half_t Bs[BN * APITCH];
  int tid  = threadIdx.x;
  int wave = tid >> 5;
  int wr = wave >> 1, wc = wave & 1;
  int m0 = blockIdx.y * BM;
  int n0 = blockIdx.x * BN;
  v8f c0 = {}, c1 = {};

  int srow = tid >> 2, scol = (tid & 3) * 8;  // 64 rows x 4 chunks of 8 halves

  const half_t* pA = &A[(long)(m0 + srow) * lda + scol];
  const half_t* pB = &Bt[(long)(n0 + srow) * ldbt + scol];

  for (int k0 = 0; k0 < K; k0 += BK) {
    *(v8h*)&As[srow * APITCH + scol] = *(const v8h*)(pA + k0);
    *(v8h*)&Bs[srow * APITCH + scol] = *(const v8h*)(pB + k0);
    __builtin_prefetch(pA + k0 + BK, 0, 0);
    __builtin_prefetch(pB + k0 + BK, 0, 0);
    __syncthreads();
    v16h a  = load_a_frag(&As[(wr * 16) * APITCH], APITCH);
    v16h b0 = load_b_frag_rows(&Bs[(wc * 32 + 0) * APITCH],  APITCH);
    v16h b1 = load_b_frag_rows(&Bs[(wc * 32 + 16) * APITCH], APITCH);
    c0 = __builtin_amdgcn_wmma_f32_16x16x32_f16(false, a, false, b0, (short)0, c0, false, false);
    c1 = __builtin_amdgcn_wmma_f32_16x16x32_f16(false, a, false, b1, (short)0, c1, false, false);
    __syncthreads();
  }

  int l = tid & 31;
  int col = l & 15;
  int rb  = (l < 16) ? 0 : 8;
#pragma unroll
  for (int r = 0; r < 8; ++r) {
    int m  = m0 + wr * 16 + rb + r;
    int nA = n0 + wc * 32 + col;
    int nB = nA + 16;
    float v0 = c0[r], v1 = c1[r];
    if (HAS_BIAS) { v0 += bias[nA]; v1 += bias[nB]; }
    if (HAS_RES)  { v0 += resid[(long)m * ldr + nA]; v1 += resid[(long)m * ldr + nB]; }
    if (GELU)     { v0 = gelu_f(v0); v1 = gelu_f(v1); }
    if (OUTF) { outF[(long)m * ldo + nA] = v0;  outF[(long)m * ldo + nB] = v1; }
    if (OUTH) { outH[(long)m * ldoh + nA] = (half_t)v0; outH[(long)m * ldoh + nB] = (half_t)v1; }
  }
}

// ---------------------------------------------------------------------------
// V transpose: qkv f16 [B*S, 3*D] -> Vt f16 [B*H*DH, S]
// ---------------------------------------------------------------------------
__global__ __launch_bounds__(256) void vtrans_kernel(const half_t* __restrict__ qkv,
                                                     half_t* __restrict__ vtm) {
  long idx = (long)blockIdx.x * 256 + threadIdx.x;  // B*H*DH*S = 2,097,152
  int  s  = (int)(idx & (S_ - 1));
  long t  = idx >> 10;
  int  d  = (int)(t & (DH_ - 1));
  int  bh = (int)(t >> 6);
  int  b  = bh >> 4, h = bh & 15;
  vtm[idx] = qkv[(long)(b * S_ + s) * (3 * D_) + (2 * H_ + h) * DH_ + d];
}

// ---------------------------------------------------------------------------
// Flash attention: one wave per 16 query rows. scores = QK^T*scale + bias +
// causal (+ window if local). Online softmax, P*V via WMMA. Out f16 [B*S, D].
// ---------------------------------------------------------------------------
__global__ __launch_bounds__(256) void flash_attn_kernel(
    const half_t* __restrict__ qkv,  // [B*S, 3*D]
    const half_t* __restrict__ vt,   // [B*H*DH, S]
    const float*  __restrict__ tb,   // [B,H,S,S]
    half_t*       __restrict__ out,  // [B*S, D]
    int local) {
  __shared__ half_t P[8][16 * APITCH];
  int wave = threadIdx.x >> 5;
  int l    = threadIdx.x & 31;
  int q0   = (blockIdx.x * 8 + wave) * 16;
  int bh   = blockIdx.y;
  int b    = bh >> 4, h = bh & 15;

  const half_t* Qb = qkv + (long)(b * S_) * (3 * D_) + h * DH_;
  const half_t* Kb = qkv + (long)(b * S_) * (3 * D_) + (H_ + h) * DH_;
  const half_t* Vb = vt  + (long)(bh * DH_) * S_;
  const float*  Bb = tb  + ((long)bh * S_ + q0) * S_;

  v16h aq0 = load_a_frag(Qb + (long)q0 * (3 * D_) + 0,  3 * D_);
  v16h aq1 = load_a_frag(Qb + (long)q0 * (3 * D_) + 32, 3 * D_);

  v8f o0 = {}, o1 = {}, o2 = {}, o3 = {};
  float mrow[8], lrow[8];
#pragma unroll
  for (int r = 0; r < 8; ++r) { mrow[r] = -3.0e38f; lrow[r] = 0.f; }

  int col = l & 15;
  int rb  = (l < 16) ? 0 : 8;
  int kend   = ((q0 + 16) + 31) & ~31;
  int kstart = 0;
  if (local) { int ks = q0 - 128; if (ks < 0) ks = 0; kstart = ks & ~31; }

  for (int kc = kstart; kc < kend; kc += 32) {
    v16h bk00 = load_b_frag_rows(Kb + (long)(kc)      * (3 * D_) + 0,  3 * D_);
    v16h bk01 = load_b_frag_rows(Kb + (long)(kc)      * (3 * D_) + 32, 3 * D_);
    v16h bk10 = load_b_frag_rows(Kb + (long)(kc + 16) * (3 * D_) + 0,  3 * D_);
    v16h bk11 = load_b_frag_rows(Kb + (long)(kc + 16) * (3 * D_) + 32, 3 * D_);
    v8f s0 = {}, s1 = {};
    s0 = __builtin_amdgcn_wmma_f32_16x16x32_f16(false, aq0, false, bk00, (short)0, s0, false, false);
    s0 = __builtin_amdgcn_wmma_f32_16x16x32_f16(false, aq1, false, bk01, (short)0, s0, false, false);
    s1 = __builtin_amdgcn_wmma_f32_16x16x32_f16(false, aq0, false, bk10, (short)0, s1, false, false);
    s1 = __builtin_amdgcn_wmma_f32_16x16x32_f16(false, aq1, false, bk11, (short)0, s1, false, false);

    float v0[8], v1[8];
#pragma unroll
    for (int r = 0; r < 8; ++r) {
      int m = rb + r, q = q0 + m;
      int nA = kc + col, nB = nA + 16;
      float x0 = s0[r] * 0.125f + Bb[(long)m * S_ + nA];
      float x1 = s1[r] * 0.125f + Bb[(long)m * S_ + nB];
      if (nA > q) x0 += NEG_;
      if (nB > q) x1 += NEG_;
      if (local) {
        int dA = q - nA; if (dA < 0) dA = -dA;
        int dB = q - nB; if (dB < 0) dB = -dB;
        if (dA > 128) x0 += NEG_;
        if (dB > 128) x1 += NEG_;
      }
      v0[r] = x0; v1[r] = x1;
    }
#pragma unroll
    for (int r = 0; r < 8; ++r) {
      float cm   = redmax16(fmaxf(v0[r], v1[r]));
      float mnew = fmaxf(mrow[r], cm);
      float sc   = __expf(mrow[r] - mnew);
      float p0   = __expf(v0[r] - mnew);
      float p1   = __expf(v1[r] - mnew);
      float rs   = redsum16(p0 + p1);
      lrow[r] = lrow[r] * sc + rs;
      mrow[r] = mnew;
      o0[r] *= sc; o1[r] *= sc; o2[r] *= sc; o3[r] *= sc;
      int m = rb + r;
      P[wave][m * APITCH + col]      = (half_t)p0;
      P[wave][m * APITCH + 16 + col] = (half_t)p1;
    }
    v16h ap  = load_a_frag(&P[wave][0], APITCH);
    v16h bv0 = load_b_frag_rows(Vb + (long)(0 * 16) * S_ + kc, S_);
    v16h bv1 = load_b_frag_rows(Vb + (long)(1 * 16) * S_ + kc, S_);
    v16h bv2 = load_b_frag_rows(Vb + (long)(2 * 16) * S_ + kc, S_);
    v16h bv3 = load_b_frag_rows(Vb + (long)(3 * 16) * S_ + kc, S_);
    o0 = __builtin_amdgcn_wmma_f32_16x16x32_f16(false, ap, false, bv0, (short)0, o0, false, false);
    o1 = __builtin_amdgcn_wmma_f32_16x16x32_f16(false, ap, false, bv1, (short)0, o1, false, false);
    o2 = __builtin_amdgcn_wmma_f32_16x16x32_f16(false, ap, false, bv2, (short)0, o2, false, false);
    o3 = __builtin_amdgcn_wmma_f32_16x16x32_f16(false, ap, false, bv3, (short)0, o3, false, false);
  }

#pragma unroll
  for (int r = 0; r < 8; ++r) {
    int m = rb + r, q = q0 + m;
    half_t* orow = out + (long)(b * S_ + q) * D_ + h * DH_;
    float inv = 1.0f / lrow[r];
    orow[0 * 16 + col] = (half_t)(o0[r] * inv);
    orow[1 * 16 + col] = (half_t)(o1[r] * inv);
    orow[2 * 16 + col] = (half_t)(o2[r] * inv);
    orow[3 * 16 + col] = (half_t)(o3[r] * inv);
  }
}

// ---------------------------------------------------------------------------
// Head tail: [2048,512] f16 @ [512,7] + b -> raw f32 [2048,7]
// ---------------------------------------------------------------------------
__global__ void head2_kernel(const half_t* __restrict__ g, const float* __restrict__ w2,
                             const float* __restrict__ b2, float* __restrict__ raw) {
  int i = blockIdx.x, j = threadIdx.x;
  if (j < 7) {
    const half_t* row = g + (long)i * (D_ / 2);
    float acc = b2[j];
    for (int k = 0; k < D_ / 2; ++k) acc += (float)row[k] * w2[k * 7 + j];
    raw[i * 7 + j] = acc;
  }
}

// learned gate: sigmoid(h . gate_w + gate_b), per row
__global__ __launch_bounds__(256) void gate_kernel(const float* __restrict__ h,
                                                   const float* __restrict__ gw,
                                                   const float* __restrict__ gb,
                                                   float* __restrict__ learned) {
  __shared__ float buf[8];
  int i = blockIdx.x;
  const float* row = h + (long)i * D_;
  float s = 0.f;
  for (int k = threadIdx.x; k < D_; k += 256) s += row[k] * gw[k];
#pragma unroll
  for (int m = 1; m < 32; m <<= 1) s += __shfl_xor(s, m, 32);
  if ((threadIdx.x & 31) == 0) buf[threadIdx.x >> 5] = s;
  __syncthreads();
  if (threadIdx.x == 0) {
    float t = 0.f;
    for (int k = 0; k < 8; ++k) t += buf[k];
    learned[i] = sigm(t + gb[0]);
  }
}

// final assembly: out row = [h(1024) | 7 scalars | gate]
__global__ __launch_bounds__(256) void final_kernel(const float* __restrict__ h,
                                                    const float* __restrict__ raw,
                                                    const float* __restrict__ learned,
                                                    const float* __restrict__ gcw,
                                                    const float* __restrict__ gcb,
                                                    float* __restrict__ outp) {
  int i = blockIdx.x;
  const float* hrow = h + (long)i * D_;
  float* orow = outp + (long)i * (D_ + 8);
  for (int t = threadIdx.x; t < D_; t += 256) orow[t] = hrow[t];
  if (threadIdx.x == 0) {
    float sc[7];
    sc[0] = sigm(raw[i * 7 + 0]); sc[1] = sigm(raw[i * 7 + 1]);
    sc[2] = tanhf(raw[i * 7 + 2]) * 2.0f;
    sc[3] = sigm(raw[i * 7 + 3]); sc[4] = sigm(raw[i * 7 + 4]);
    sc[5] = sigm(raw[i * 7 + 5]); sc[6] = sigm(raw[i * 7 + 6]);
#pragma unroll
    for (int j = 0; j < 7; ++j) orow[D_ + j] = sc[j];
    orow[D_ + 7] = sigm(gcw[0] * learned[i] + gcw[1] * sc[0] + gcb[0]);
  }
}

// ---------------------------------------------------------------------------
// Host launcher
// ---------------------------------------------------------------------------
extern "C" void kernel_launch(void* const* d_in, const int* in_sizes, int n_in,
                              void* d_out, int out_size, void* d_ws, size_t ws_size,
                              hipStream_t stream) {
  (void)in_sizes; (void)n_in; (void)out_size; (void)ws_size;
  const float* x       = (const float*)d_in[0];
  const float* tb      = (const float*)d_in[1];
  const float* qkv_w   = (const float*)d_in[2];
  const float* out_w   = (const float*)d_in[3];
  const float* w1      = (const float*)d_in[4];
  const float* b1      = (const float*)d_in[5];
  const float* w2      = (const float*)d_in[6];
  const float* b2      = (const float*)d_in[7];
  const float* ln1_g   = (const float*)d_in[8];
  const float* ln1_b   = (const float*)d_in[9];
  const float* ln2_g   = (const float*)d_in[10];
  const float* ln2_b   = (const float*)d_in[11];
  const float* hln_g   = (const float*)d_in[12];
  const float* hln_b   = (const float*)d_in[13];
  const float* hw1     = (const float*)d_in[14];
  const float* hb1     = (const float*)d_in[15];
  const float* hw2     = (const float*)d_in[16];
  const float* hb2     = (const float*)d_in[17];
  const float* gate_w  = (const float*)d_in[18];
  const float* gate_b  = (const float*)d_in[19];
  const float* gatec_w = (const float*)d_in[20];
  const float* gatec_b = (const float*)d_in[21];

  const long M = (long)B_ * S_;  // 2048
  char* ws = (char*)d_ws;
  // workspace layout (~58 MB total)
  float*  h       = (float*)(ws);                       // 8 MB
  half_t* yf16    = (half_t*)(ws + 8388608);            // 4 MB
  half_t* wf16    = (half_t*)(ws + 12582912);           // 8 MB (max weight, [N,K])
  half_t* qkvf16  = (half_t*)(ws + 20971520);           // 12 MB
  half_t* vtf16   = (half_t*)(ws + 33554432);           // 4 MB
  half_t* attnf16 = (half_t*)(ws + 37748736);           // 4 MB
  half_t* midf16  = (half_t*)(ws + 41943040);           // 16 MB
  half_t* gf16    = (half_t*)(ws + 58720256);           // 2 MB
  float*  raw     = (float*)(ws + 60817408);            // 56 KB
  float*  learned = (float*)(ws + 60874752);            // 8 KB

  hipMemcpyAsync(h, x, M * D_ * sizeof(float), hipMemcpyDeviceToDevice, stream);

  for (int l = 0; l < L_; ++l) {
    // --- attention block ---
    ln_kernel<<<M, 256, 0, stream>>>(h, ln1_g + l * D_, ln1_b + l * D_, yf16);
    cvtT_kernel<<<dim3(3 * D_ / 32, D_ / 32), 256, 0, stream>>>(
        qkv_w + (long)l * D_ * 3 * D_, wf16, D_, 3 * D_);
    gemm_kernel<false, false, false, false, true>
        <<<dim3(3 * D_ / BN, M / BM), 256, 0, stream>>>(
            yf16, D_, wf16, D_, nullptr, nullptr, 0, nullptr, 0,
            qkvf16, 3 * D_, (int)M, 3 * D_, D_);
    vtrans_kernel<<<(B_ * H_ * DH_ * S_) / 256, 256, 0, stream>>>(qkvf16, vtf16);
    flash_attn_kernel<<<dim3(S_ / 128, B_ * H_), 256, 0, stream>>>(
        qkvf16, vtf16, tb, attnf16, (l % 2 == 0) ? 1 : 0);
    cvtT_kernel<<<dim3(D_ / 32, D_ / 32), 256, 0, stream>>>(
        out_w + (long)l * D_ * D_, wf16, D_, D_);
    gemm_kernel<false, true, false, true, false>
        <<<dim3(D_ / BN, M / BM), 256, 0, stream>>>(
            attnf16, D_, wf16, D_, nullptr, h, D_, h, D_, nullptr, 0,
            (int)M, D_, D_);
    // --- FFN block ---
    ln_kernel<<<M, 256, 0, stream>>>(h, ln2_g + l * D_, ln2_b + l * D_, yf16);
    cvtT_kernel<<<dim3(FF_ / 32, D_ / 32), 256, 0, stream>>>(
        w1 + (long)l * D_ * FF_, wf16, D_, FF_);
    gemm_kernel<true, false, true, false, true>
        <<<dim3(FF_ / BN, M / BM), 256, 0, stream>>>(
            yf16, D_, wf16, D_, b1 + (long)l * FF_, nullptr, 0, nullptr, 0,
            midf16, FF_, (int)M, FF_, D_);
    cvtT_kernel<<<dim3(D_ / 32, FF_ / 32), 256, 0, stream>>>(
        w2 + (long)l * FF_ * D_, wf16, FF_, D_);
    gemm_kernel<true, true, false, true, false>
        <<<dim3(D_ / BN, M / BM), 256, 0, stream>>>(
            midf16, FF_, wf16, FF_, b2 + (long)l * D_, h, D_, h, D_, nullptr, 0,
            (int)M, D_, FF_);
  }

  // --- head ---
  ln_kernel<<<M, 256, 0, stream>>>(h, hln_g, hln_b, yf16);
  cvtT_kernel<<<dim3((D_ / 2) / 32, D_ / 32), 256, 0, stream>>>(
      hw1, wf16, D_, D_ / 2);
  gemm_kernel<true, false, true, false, true>
      <<<dim3((D_ / 2) / BN, M / BM), 256, 0, stream>>>(
          yf16, D_, wf16, D_, hb1, nullptr, 0, nullptr, 0,
          gf16, D_ / 2, (int)M, D_ / 2, D_);
  head2_kernel<<<M, 32, 0, stream>>>(gf16, hw2, hb2, raw);
  gate_kernel<<<M, 256, 0, stream>>>(h, gate_w, gate_b, learned);
  final_kernel<<<M, 256, 0, stream>>>(h, raw, learned, gatec_w, gatec_b, (float*)d_out);
}
